// MHGCN_26147760898472
// MI455X (gfx1250) — matrix operand order repaired and checked by reading, not compiled, not run on hardware
//
#include <hip/hip_runtime.h>
#include <hip/hip_bf16.h>

#define NN     8192
#define OUTF   128
#define KSPLIT 4
#define KCHUNK (NN / KSPLIT)   // 2048

typedef __attribute__((ext_vector_type(16))) __bf16 v16bf;
typedef __attribute__((ext_vector_type(8)))  float  v8f;

// ---------------------------------------------------------------------------
// Kernel 1: adj = (A[:,:,0]*r0 + A[:,:,1]*r1); adj = adj + adj^T, stored bf16.
// Block handles tile pair (bi,bj)/(bj,bi) with bi<=bj -> each A element read once.
// ---------------------------------------------------------------------------
__global__ void __launch_bounds__(256)
combine_sym(const float2* __restrict__ A2, const float* __restrict__ rw,
            __bf16* __restrict__ adj) {
  __shared__ float s1[64][65];
  __shared__ float s2[64][65];
  const int bi = blockIdx.y, bj = blockIdx.x;
  if (bj < bi) return;
  const float r0 = rw[0], r1 = rw[1];
  const int t = threadIdx.x;
#pragma unroll
  for (int p = 0; p < 16; ++p) {
    int idx = t + p * 256;
    int r = idx >> 6, c = idx & 63;
    float2 v1 = A2[(size_t)(bi * 64 + r) * NN + (bj * 64 + c)];
    float2 v2 = A2[(size_t)(bj * 64 + r) * NN + (bi * 64 + c)];
    s1[r][c] = r0 * v1.x + r1 * v1.y;
    s2[r][c] = r0 * v2.x + r1 * v2.y;
  }
  __syncthreads();
#pragma unroll
  for (int p = 0; p < 16; ++p) {
    int idx = t + p * 256;
    int r = idx >> 6, c = idx & 63;
    adj[(size_t)(bi * 64 + r) * NN + (bj * 64 + c)] = (__bf16)(s1[r][c] + s2[c][r]);
    if (bi != bj)
      adj[(size_t)(bj * 64 + r) * NN + (bi * 64 + c)] = (__bf16)(s2[r][c] + s1[c][r]);
  }
}

// ---------------------------------------------------------------------------
// Kernel 2/5: Gt[n, i] = sum_k W[k, n] * S[i, k]   (i.e. (S @ W)^T), bf16 out.
// WMMA with A = W^T (16x32), B = S^T (32x16); one wave per 16x16 output tile.
// ---------------------------------------------------------------------------
__global__ void __launch_bounds__(256)
gemm_t_small(const float* __restrict__ W, const float* __restrict__ S,
             __bf16* __restrict__ Gt, int K) {
  const int wave = (int)((blockIdx.x * blockDim.x + threadIdx.x) >> 5);
  const int lane = threadIdx.x & 31;
  const int half = lane >> 4, mrow = lane & 15;
  const int n0 = (wave & 7) * 16;          // output-feature tile (M dim of WMMA)
  const int i0 = (wave >> 3) * 16;         // node tile (N dim of WMMA)
  v8f acc = {};
  for (int k0 = 0; k0 < K; k0 += 32) {
    v16bf a, b;
#pragma unroll
    for (int e = 0; e < 8; ++e) {
      a[e]     = (__bf16)W[(size_t)(k0 + half * 16 + e)      * OUTF + n0 + mrow];
      a[e + 8] = (__bf16)W[(size_t)(k0 + half * 16 + 8 + e) * OUTF + n0 + mrow];
    }
#pragma unroll
    for (int e = 0; e < 16; ++e)
      b[e] = (__bf16)S[(size_t)(i0 + mrow) * K + k0 + half * 16 + e];
    acc = __builtin_amdgcn_wmma_f32_16x16x32_bf16(false, a, false, b,
                                                  (short)0, acc, false, false);
  }
#pragma unroll
  for (int r = 0; r < 8; ++r)
    Gt[(size_t)(n0 + half * 8 + r) * NN + i0 + mrow] = (__bf16)acc[r];
}

// ---------------------------------------------------------------------------
// Kernel 3/6: k-split aggregation GEMM. Wave w: strip = w & 511 (16 rows),
// split = w >> 9 (k-range of 2048). 8 accumulator tiles cover all 128 cols, so
// each adj A-fragment feeds 8 WMMAs; adj is read exactly once per layer.
// Writes f32 partial slice partial[split][8192][128].
// ---------------------------------------------------------------------------
__global__ void __launch_bounds__(256)
agg_gemm_split(const __bf16* __restrict__ adj, const __bf16* __restrict__ Gt,
               float* __restrict__ partial) {
  const int wave  = (int)((blockIdx.x * blockDim.x + threadIdx.x) >> 5);
  const int lane  = threadIdx.x & 31;
  const int half  = lane >> 4, mrow = lane & 15;
  const int strip = wave & 511;
  const int split = wave >> 9;
  const int row0  = strip * 16;
  const int ks    = split * KCHUNK;
  const int kend  = ks + KCHUNK;

  v8f acc[8] = {};
  const __bf16* pa  = adj + (size_t)(row0 + mrow) * NN + half * 8;
  const __bf16* pb0 = Gt  + (size_t)mrow * NN + half * 8;

  for (int k0 = ks; k0 < kend; k0 += 32) {
    if (k0 + 1024 < kend) __builtin_prefetch(pa + k0 + 1024, 0, 0);
    v16bf a;
    *(uint4*)&a       = *(const uint4*)(pa + k0);
    *((uint4*)&a + 1) = *(const uint4*)(pa + k0 + 16);
#pragma unroll
    for (int t = 0; t < 8; ++t) {
      v16bf b;
      const __bf16* pb = pb0 + (size_t)t * 16 * NN + k0;
      *(uint4*)&b       = *(const uint4*)(pb);
      *((uint4*)&b + 1) = *(const uint4*)(pb + 16);
      acc[t] = __builtin_amdgcn_wmma_f32_16x16x32_bf16(false, a, false, b,
                                                       (short)0, acc[t], false, false);
    }
  }

  float* ps = partial + (size_t)split * NN * OUTF;
#pragma unroll
  for (int t = 0; t < 8; ++t) {
    const int n = t * 16 + mrow;
#pragma unroll
    for (int r = 0; r < 8; ++r) {
      const int i = row0 + half * 8 + r;
      ps[(size_t)i * OUTF + n] = acc[t][r];
    }
  }
}

// ---------------------------------------------------------------------------
// Kernel 4/7: finalize. out[i,n] = sum_s partial[s][i,n] + bias[n];
// if prev: out = 0.5 * (prev + out).
// ---------------------------------------------------------------------------
__global__ void __launch_bounds__(256)
finalize(const float* __restrict__ partial, const float* __restrict__ bias,
         const float* __restrict__ prev, float* __restrict__ out) {
  const int idx = blockIdx.x * 256 + threadIdx.x;   // 0 .. 8192*128-1
  const int n = idx & (OUTF - 1);
  float v = bias[n];
#pragma unroll
  for (int s = 0; s < KSPLIT; ++s)
    v += partial[(size_t)s * NN * OUTF + idx];
  if (prev) v = 0.5f * (prev[idx] + v);
  out[idx] = v;
}

// ---------------------------------------------------------------------------
extern "C" void kernel_launch(void* const* d_in, const int* in_sizes, int n_in,
                              void* d_out, int out_size, void* d_ws, size_t ws_size,
                              hipStream_t stream) {
  const float* x  = (const float*)d_in[0];   // [8192, 256]
  const float* A  = (const float*)d_in[1];   // [8192, 8192, 2]
  const float* w0 = (const float*)d_in[2];   // [256, 128]
  const float* b0 = (const float*)d_in[3];   // [128]
  const float* w1 = (const float*)d_in[4];   // [128, 128]
  const float* b1 = (const float*)d_in[5];   // [128]
  const float* rw = (const float*)d_in[6];   // [2, 1]

  char* ws = (char*)d_ws;
  const size_t ADJ_BYTES = (size_t)NN * NN * sizeof(__bf16);        // 134 MB
  const size_t GT_BYTES  = (size_t)OUTF * NN * sizeof(__bf16);      // 2 MB
  const size_t H_BYTES   = (size_t)NN * OUTF * sizeof(float);       // 4 MB
  __bf16* adj  = (__bf16*)(ws);
  __bf16* G0t  = (__bf16*)(ws + ADJ_BYTES);
  __bf16* G1t  = (__bf16*)(ws + ADJ_BYTES + GT_BYTES);
  float*  h1f  = (float*) (ws + ADJ_BYTES + 2 * GT_BYTES);
  float*  part = (float*) (ws + ADJ_BYTES + 2 * GT_BYTES + H_BYTES); // 16 MB

  const int AGG_BLOCKS = (512 * KSPLIT) / 8;   // 2048 waves, 8 waves/block
  const int FIN_BLOCKS = (NN * OUTF) / 256;    // 4096

  // 1) adj (bf16, symmetric) from relation mix — reads A exactly once.
  combine_sym<<<dim3(128, 128), 256, 0, stream>>>((const float2*)A, rw, adj);
  // 2) G0^T = (x @ w0)^T   [128, 8192] bf16
  gemm_t_small<<<512, 256, 0, stream>>>(w0, x, G0t, 256);
  // 3) partial = adj @ G0 (k-split x4)
  agg_gemm_split<<<AGG_BLOCKS, 256, 0, stream>>>(adj, G0t, part);
  // 4) h1 = sum(partial) + b0
  finalize<<<FIN_BLOCKS, 256, 0, stream>>>(part, b0, nullptr, h1f);
  // 5) G1^T = (h1 @ w1)^T  [128, 8192] bf16
  gemm_t_small<<<512, 256, 0, stream>>>(w1, h1f, G1t, 128);
  // 6) partial = adj @ G1 (k-split x4)
  agg_gemm_split<<<AGG_BLOCKS, 256, 0, stream>>>(adj, G1t, part);
  // 7) out = 0.5 * (h1 + partial-sum + b1)
  finalize<<<FIN_BLOCKS, 256, 0, stream>>>(part, b1, h1f, (float*)d_out);
}
